// MultiTaskMOE_37503654429111
// MI455X (gfx1250) — matrix operand order repaired and checked by reading, not compiled, not run on hardware
//
#include <hip/hip_runtime.h>
#include <cstddef>
#include <cstdint>

// ---------------------------------------------------------------------------
// MultiTask MoE forward for MI455X (gfx1250, wave32, WMMA).
// 6 dense expert FFNs over 8192 tokens. GEMMs use bf16x2-split WMMA
// (hi*hi + hi*lo + lo*hi) for ~fp32 accuracy on the bf16 tensor path.
// Block tile 128x128, 256 threads (8 waves), wave tile 32x64 = 2x4 WMMA
// subtiles -> 24 v_wmma per wave per K=32 chunk, ds:wmma issue ratio ~1:1.
// ---------------------------------------------------------------------------

typedef __attribute__((ext_vector_type(16))) __bf16 v16bf;
typedef __attribute__((ext_vector_type(8)))  float  v8f;
typedef __attribute__((ext_vector_type(2)))  __bf16 bf16x2;

#define BM 128
#define BN 128
#define TK 32
#define LP 40           // padded LDS K-stride (20-bank row stride, conflict-free)
#define NTHR 256

#define DD 1024         // model dim
#define HH 4096         // hidden dim
#define MTOK 8192       // B*N tokens
#define NSLOT 6         // gate outputs per task
#define NTASK 2

struct Tiles {
  __bf16 Ah[BM][LP];
  __bf16 Al[BM][LP];
  __bf16 Bh[BN][LP];    // stored transposed: [n][k]
  __bf16 Bl[BN][LP];
};

// ---- WMMA fragment loaders (layouts per CDNA5 ISA 7.12.2) -----------------
// A 16x32 bf16: lane l -> m = l&15, kbase = (l>>4)*8;
//   VGPR j holds k = kbase + (j>>2)*16 + (j&3)*2 (pair k,k+1)
__device__ __forceinline__ v16bf frag_a(const __bf16 (*T)[LP], int rowbase, int lane) {
  const int m  = rowbase + (lane & 15);
  const int kb = (lane >> 4) << 3;
  v16bf f;
#pragma unroll
  for (int j = 0; j < 8; ++j) {
    const int k = kb + ((j >> 2) << 4) + ((j & 3) << 1);
    bf16x2 p = *(const bf16x2*)&T[m][k];
    f[2 * j] = p[0];
    f[2 * j + 1] = p[1];
  }
  return f;
}

// B 32x16 bf16: lane l -> n = l&15, kbase = (l>>4)*8; VGPR j holds k = kbase + (j>>2)*16 + (j&3)*2
__device__ __forceinline__ v16bf frag_b(const __bf16 (*T)[LP], int colbase, int lane) {
  const int n  = colbase + (lane & 15);
  const int kb = (lane >> 4) << 3;
  v16bf f;
#pragma unroll
  for (int j = 0; j < 8; ++j) {
    const int k = kb + ((j >> 2) << 4) + ((j & 3) << 1);
    bf16x2 p = *(const bf16x2*)&T[n][k];
    f[2 * j] = p[0];
    f[2 * j + 1] = p[1];
  }
  return f;
}

__device__ __forceinline__ void split_store(float v, __bf16* hi, __bf16* lo) {
  __bf16 h = (__bf16)v;
  *hi = h;
  *lo = (__bf16)(v - (float)h);
}

// ---- 128x128 block tile GEMM core (A fp32 MxK, B fp32 KxN) ----------------
// 256 threads = 8 waves in 4x2; each wave owns a 32x64 region = 2x4 WMMA tiles.
__device__ __forceinline__ void mm_core(const float* __restrict__ A, int lda,
                                        const float* __restrict__ B, int ldb,
                                        int K, int m0, int n0,
                                        Tiles* sm, v8f acc[2][4]) {
  const int t    = threadIdx.x;
  const int lane = t & 31;
  const int wv   = t >> 5;
  const int wm   = (wv >> 1) << 5;   // wave row offset in block tile (0/32/64/96)
  const int wn   = (wv & 1) << 6;    // wave col offset (0/64)

  for (int kc = 0; kc < K; kc += TK) {
    // A tile: 128 rows x 32 k (fp32 -> split bf16 hi/lo)
#pragma unroll
    for (int i = 0; i < 4; ++i) {
      const int row = (t >> 3) + (i << 5);
      const int col = (t & 7) << 2;
      const float4 v = *(const float4*)(A + (size_t)(m0 + row) * lda + kc + col);
      split_store(v.x, &sm->Ah[row][col + 0], &sm->Al[row][col + 0]);
      split_store(v.y, &sm->Ah[row][col + 1], &sm->Al[row][col + 1]);
      split_store(v.z, &sm->Ah[row][col + 2], &sm->Al[row][col + 2]);
      split_store(v.w, &sm->Ah[row][col + 3], &sm->Al[row][col + 3]);
    }
    // B tile: 32 k-rows x 128 n, stored transposed [n][k]
#pragma unroll
    for (int i = 0; i < 4; ++i) {
      const int kr = (t >> 5) + (i << 3);
      const int nc = (t & 31) << 2;
      const float4 v = *(const float4*)(B + (size_t)(kc + kr) * ldb + n0 + nc);
      split_store(v.x, &sm->Bh[nc + 0][kr], &sm->Bl[nc + 0][kr]);
      split_store(v.y, &sm->Bh[nc + 1][kr], &sm->Bl[nc + 1][kr]);
      split_store(v.z, &sm->Bh[nc + 2][kr], &sm->Bl[nc + 2][kr]);
      split_store(v.w, &sm->Bh[nc + 3][kr], &sm->Bl[nc + 3][kr]);
    }
    __syncthreads();

    // Prefetch next K tile into caches (global_prefetch_b8 on gfx1250)
    if (kc + TK < K) {
      __builtin_prefetch(A + (size_t)(m0 + (t >> 3)) * lda + kc + TK, 0, 1);
      __builtin_prefetch(B + (size_t)(kc + TK + (t >> 5)) * ldb + n0 + ((t & 31) << 2), 0, 1);
    }

    v16bf ah[2], al[2], bh[4], bl[4];
#pragma unroll
    for (int mi = 0; mi < 2; ++mi) {
      ah[mi] = frag_a(sm->Ah, wm + (mi << 4), lane);
      al[mi] = frag_a(sm->Al, wm + (mi << 4), lane);
    }
#pragma unroll
    for (int ni = 0; ni < 4; ++ni) {
      bh[ni] = frag_b(sm->Bh, wn + (ni << 4), lane);
      bl[ni] = frag_b(sm->Bl, wn + (ni << 4), lane);
    }
#pragma unroll
    for (int mi = 0; mi < 2; ++mi) {
#pragma unroll
      for (int ni = 0; ni < 4; ++ni) {
        acc[mi][ni] = __builtin_amdgcn_wmma_f32_16x16x32_bf16(
            false, ah[mi], false, bh[ni], (short)0, acc[mi][ni], false, false);
        acc[mi][ni] = __builtin_amdgcn_wmma_f32_16x16x32_bf16(
            false, ah[mi], false, bl[ni], (short)0, acc[mi][ni], false, false);
        acc[mi][ni] = __builtin_amdgcn_wmma_f32_16x16x32_bf16(
            false, al[mi], false, bh[ni], (short)0, acc[mi][ni], false, false);
      }
    }
    __syncthreads();
  }
}

__device__ __forceinline__ void acc_zero(v8f acc[2][4]) {
#pragma unroll
  for (int mi = 0; mi < 2; ++mi)
#pragma unroll
    for (int ni = 0; ni < 4; ++ni)
#pragma unroll
      for (int r = 0; r < 8; ++r) acc[mi][ni][r] = 0.0f;
}

// ---- GEMM1: C = gelu_exact(A*B + bias) ------------------------------------
__global__ __launch_bounds__(NTHR)
void gemm_gelu_kernel(const float* __restrict__ A, int lda,
                      const float* __restrict__ B, int ldb,
                      const float* __restrict__ bias,
                      float* __restrict__ C, int ldc, int K) {
  __shared__ Tiles sm;
  v8f acc[2][4];
  acc_zero(acc);
  const int m0 = blockIdx.x * BM, n0 = blockIdx.y * BN;
  mm_core(A, lda, B, ldb, K, m0, n0, &sm, acc);

  const int lane = threadIdx.x & 31, wv = threadIdx.x >> 5;
  const int wm = (wv >> 1) << 5, wn = (wv & 1) << 6;
#pragma unroll
  for (int mi = 0; mi < 2; ++mi) {
#pragma unroll
    for (int ni = 0; ni < 4; ++ni) {
      const int mb = m0 + wm + (mi << 4) + ((lane >> 4) << 3);  // + r below
      const int nb = n0 + wn + (ni << 4) + (lane & 15);
      const float bz = bias[nb];
#pragma unroll
      for (int r = 0; r < 8; ++r) {
        float v = acc[mi][ni][r] + bz;
        v = 0.5f * v * (1.0f + erff(v * 0.70710678118654752f));
        C[(size_t)(mb + r) * ldc + nb] = v;
      }
    }
  }
}

// ---- GEMM2: out[t] += gate_w[t] * (A*B + bias) (up to two tasks) ----------
__global__ __launch_bounds__(NTHR)
void gemm_moe_kernel(const float* __restrict__ A, int lda,
                     const float* __restrict__ B, int ldb,
                     const float* __restrict__ bias, int K,
                     float* __restrict__ outA, const float* __restrict__ gwA,
                     float* __restrict__ outB, const float* __restrict__ gwB,
                     int ldc) {
  __shared__ Tiles sm;
  v8f acc[2][4];
  acc_zero(acc);
  const int m0 = blockIdx.x * BM, n0 = blockIdx.y * BN;
  mm_core(A, lda, B, ldb, K, m0, n0, &sm, acc);

  const int lane = threadIdx.x & 31, wv = threadIdx.x >> 5;
  const int wm = (wv >> 1) << 5, wn = (wv & 1) << 6;
#pragma unroll
  for (int mi = 0; mi < 2; ++mi) {
#pragma unroll
    for (int ni = 0; ni < 4; ++ni) {
      const int mb = m0 + wm + (mi << 4) + ((lane >> 4) << 3);
      const int nb = n0 + wn + (ni << 4) + (lane & 15);
      const float bz = bias[nb];
#pragma unroll
      for (int r = 0; r < 8; ++r) {
        const int m = mb + r;
        const float y = acc[mi][ni][r] + bz;
        const size_t o = (size_t)m * ldc + nb;
        outA[o] += gwA[(size_t)m * NSLOT] * y;
        if (outB) outB[o] += gwB[(size_t)m * NSLOT] * y;
      }
    }
  }
}

// ---- Gating: logits, top-2 softmax weights, aux f/P partial sums ----------
__global__ __launch_bounds__(256)
void gate_kernel(const float* __restrict__ x, const float* __restrict__ Wg,
                 float* __restrict__ wbuf, float* __restrict__ fSum,
                 float* __restrict__ pSum) {
  __shared__ float red[24];  // [0..11] f counts, [12..23] P sums
  if (threadIdx.x < 24) red[threadIdx.x] = 0.0f;
  __syncthreads();

  const int lane = threadIdx.x & 31, wv = threadIdx.x >> 5;
  const int token = blockIdx.x * 8 + wv;  // 8 waves per block, 1 token per wave

  float s[12];
#pragma unroll
  for (int j = 0; j < 12; ++j) s[j] = 0.0f;
  const float* xr = x + (size_t)token * DD;
  for (int d = lane; d < DD; d += 32) {
    const float xv = xr[d];
#pragma unroll
    for (int j = 0; j < 12; ++j) s[j] += xv * Wg[(size_t)j * DD + d];
  }
#pragma unroll
  for (int off = 16; off > 0; off >>= 1) {
#pragma unroll
    for (int j = 0; j < 12; ++j) s[j] += __shfl_down(s[j], off, 32);
  }

  if (lane == 0) {
    for (int t = 0; t < NTASK; ++t) {
      const float* lg = s + t * NSLOT;
      // top-2 (first index wins ties, matching lax.top_k)
      int i0 = 0;
      for (int j = 1; j < NSLOT; ++j)
        if (lg[j] > lg[i0]) i0 = j;
      int i1 = -1;
      for (int j = 0; j < NSLOT; ++j)
        if (j != i0 && (i1 < 0 || lg[j] > lg[i1])) i1 = j;
      const float e = expf(lg[i1] - lg[i0]);
      const float inv = 1.0f / (1.0f + e);
      float wr[NSLOT];
#pragma unroll
      for (int j = 0; j < NSLOT; ++j) wr[j] = 0.0f;
      wr[i0] = inv;
      wr[i1] = e * inv;
      float* wd = wbuf + (size_t)t * MTOK * NSLOT + (size_t)token * NSLOT;
#pragma unroll
      for (int j = 0; j < NSLOT; ++j) wd[j] = wr[j];
      // P = softmax over all 6 logits; f = one-hot counts
      float p[NSLOT], sum = 0.0f;
      const float mx = lg[i0];
#pragma unroll
      for (int j = 0; j < NSLOT; ++j) { p[j] = expf(lg[j] - mx); sum += p[j]; }
      const float rs = 1.0f / sum;
#pragma unroll
      for (int j = 0; j < NSLOT; ++j) atomicAdd(&red[12 + t * NSLOT + j], p[j] * rs);
      atomicAdd(&red[t * NSLOT + i0], 1.0f);
      atomicAdd(&red[t * NSLOT + i1], 1.0f);
    }
  }
  __syncthreads();
  if (threadIdx.x < 12)
    atomicAdd(&fSum[threadIdx.x], red[threadIdx.x]);
  else if (threadIdx.x < 24)
    atomicAdd(&pSum[threadIdx.x - 12], red[threadIdx.x]);
}

__global__ __launch_bounds__(256)
void zero_kernel(float* __restrict__ out, size_t n, float* __restrict__ aux, int na) {
  for (size_t i = (size_t)blockIdx.x * blockDim.x + threadIdx.x; i < n;
       i += (size_t)gridDim.x * blockDim.x)
    out[i] = 0.0f;
  if (blockIdx.x == 0 && (int)threadIdx.x < na) aux[threadIdx.x] = 0.0f;
}

__global__ void aux_kernel(const float* __restrict__ fSum,
                           const float* __restrict__ pSum,
                           float* __restrict__ auxOut) {
  const int t = threadIdx.x;
  if (t < NTASK) {
    float a = 0.0f;
    const float im = 1.0f / (float)MTOK;
#pragma unroll
    for (int j = 0; j < NSLOT; ++j)
      a += (fSum[t * NSLOT + j] * im) * (pSum[t * NSLOT + j] * im);
    auxOut[t] = 2.0f * a;  // S * sum(f*P), S = 2
  }
}

// ---------------------------------------------------------------------------
extern "C" void kernel_launch(void* const* d_in, const int* in_sizes, int n_in,
                              void* d_out, int out_size, void* d_ws, size_t ws_size,
                              hipStream_t stream) {
  (void)in_sizes; (void)n_in; (void)out_size; (void)ws_size;
  const float* x   = (const float*)d_in[0];
  const float* Wg  = (const float*)d_in[1];
  const float* sW1 = (const float*)d_in[2];
  const float* sb1 = (const float*)d_in[3];
  const float* sW2 = (const float*)d_in[4];
  const float* sb2 = (const float*)d_in[5];
  const float* tW1 = (const float*)d_in[6];
  const float* tb1 = (const float*)d_in[7];
  const float* tW2 = (const float*)d_in[8];
  const float* tb2 = (const float*)d_in[9];
  float* out = (float*)d_out;

  // workspace: h (8192x4096 f32) | gate weights (2x8192x6) | fSum(12) | pSum(12)
  float* h    = (float*)d_ws;
  float* wbuf = h + (size_t)MTOK * HH;
  float* fSum = wbuf + (size_t)NTASK * MTOK * NSLOT;
  float* pSum = fSum + 12;

  const size_t outElems = (size_t)NTASK * MTOK * DD;

  zero_kernel<<<2048, 256, 0, stream>>>(out, outElems, fSum, 24);
  gate_kernel<<<MTOK / 8, 256, 0, stream>>>(x, Wg, wbuf, fSum, pSum);

  float* out0 = out;
  float* out1 = out + (size_t)MTOK * DD;
  const float* w0 = wbuf;
  const float* w1 = wbuf + (size_t)MTOK * NSLOT;

  struct E { const float *W1, *b1, *W2, *b2; float* oA; const float* gA;
             float* oB; const float* gB; };
  const E ex[6] = {
    // shared experts feed both tasks at gate slots 0,1
    { sW1,                      sb1,          sW2,                      sb2,          out0, w0 + 0, out1, w1 + 0 },
    { sW1 + (size_t)DD * HH,    sb1 + HH,     sW2 + (size_t)HH * DD,    sb2 + DD,     out0, w0 + 1, out1, w1 + 1 },
    // task experts at gate slots 2,3 of their own task
    { tW1,                      tb1,          tW2,                      tb2,          out0, w0 + 2, nullptr, nullptr },
    { tW1 + (size_t)DD * HH,    tb1 + HH,     tW2 + (size_t)HH * DD,    tb2 + DD,     out0, w0 + 3, nullptr, nullptr },
    { tW1 + (size_t)2 * DD * HH, tb1 + 2 * HH, tW2 + (size_t)2 * HH * DD, tb2 + 2 * DD, out1, w1 + 2, nullptr, nullptr },
    { tW1 + (size_t)3 * DD * HH, tb1 + 3 * HH, tW2 + (size_t)3 * HH * DD, tb2 + 3 * DD, out1, w1 + 3, nullptr, nullptr },
  };

  const dim3 g1(MTOK / BM, HH / BN);  // 64 x 32
  const dim3 g2(MTOK / BM, DD / BN);  // 64 x 8
  for (int i = 0; i < 6; ++i) {
    gemm_gelu_kernel<<<g1, NTHR, 0, stream>>>(x, DD, ex[i].W1, HH, ex[i].b1, h, HH, DD);
    gemm_moe_kernel<<<g2, NTHR, 0, stream>>>(h, HH, ex[i].W2, DD, ex[i].b2, HH,
                                             ex[i].oA, ex[i].gA, ex[i].oB, ex[i].gB, DD);
  }
  aux_kernel<<<1, 32, 0, stream>>>(fSum, pSum, out + outElems);
}
